// InteractionSelfAttention_44951127719997
// MI455X (gfx1250) — matrix operand order repaired and checked by reading, not compiled, not run on hardware
//
#include <hip/hip_runtime.h>
#include <hip/hip_bf16.h>
#include <stdint.h>
#include <stddef.h>

// ---------------------------------------------------------------------------
// Types / constants
// ---------------------------------------------------------------------------
typedef __attribute__((ext_vector_type(16))) __bf16          v16bf;
typedef __attribute__((ext_vector_type(8)))  float           v8f;
typedef __attribute__((ext_vector_type(8)))  unsigned short  usv8;
typedef __attribute__((ext_vector_type(4)))  unsigned short  usv4;

#define NB 8      // batch
#define BN 2048   // sequence length
#define DD 768    // hidden dim

__device__ __forceinline__ unsigned short f2bf(float f) {
  unsigned int u = __float_as_uint(f);
  u += 0x7FFFu + ((u >> 16) & 1u);   // round-to-nearest-even
  return (unsigned short)(u >> 16);
}

union FragAB {
  v16bf bf;
  usv8  u[2];
};

__device__ __forceinline__ v8f wmma_bf16(const FragAB& a, const FragAB& b, v8f c) {
  return __builtin_amdgcn_wmma_f32_16x16x32_bf16(false, a.bf, false, b.bf,
                                                 (short)0, c, false, false);
}

// ---------------------------------------------------------------------------
// Async global->LDS staging via inline asm (gfx1250 GLOBAL_LOAD_ASYNC_TO_LDS,
// tracked by ASYNCcnt; ISA 15.18.3 opcode 98).  VDST = per-lane LDS byte
// address (low 32 bits of the flat pointer, per ISA 10.2 LDS aperture rule),
// VADDR = 64-bit global address, "off" = no SADDR.
// ---------------------------------------------------------------------------
#ifndef ATHENA_ASYNC_ASM
#define ATHENA_ASYNC_ASM 1
#endif

__device__ __forceinline__ void async_cp16(const unsigned short* g,
                                           unsigned short* l) {
#if ATHENA_ASYNC_ASM
  const unsigned int       loff = (unsigned int)(size_t)l;
  const unsigned long long ga   = (unsigned long long)(size_t)g;
  asm volatile("global_load_async_to_lds_b128 %0, %1, off"
               :
               : "v"(loff), "v"(ga)
               : "memory");
#else
  *(usv8*)l = *(const usv8*)g;
#endif
}

__device__ __forceinline__ void async_wait_all() {
#if ATHENA_ASYNC_ASM
  asm volatile("s_wait_asynccnt 0x0" ::: "memory");
#endif
}

// ---------------------------------------------------------------------------
// 1) fp32 -> bf16 conversion (weights + activations)
// ---------------------------------------------------------------------------
__global__ void cvt_f32_bf16(const float* __restrict__ in,
                             unsigned short* __restrict__ out, int n) {
  const int i = (blockIdx.x * blockDim.x + threadIdx.x) * 4;
  if (i + 3 < n) {
    const float4 f = *(const float4*)(in + i);
    usv4 h;
    h[0] = f2bf(f.x); h[1] = f2bf(f.y); h[2] = f2bf(f.z); h[3] = f2bf(f.w);
    *(usv4*)(out + i) = h;
  }
}

// ---------------------------------------------------------------------------
// 2) GEMM: Out[M,768] = A[M,768](bf16) @ Bw[768,768]^T(bf16) + bias
//    block tile 128x128, 8 waves (2x4), wave tile 64x32, K-step 32,
//    double-buffered LDS with async staging
// ---------------------------------------------------------------------------
template <bool OUT_F32>
__global__ __launch_bounds__(256) void gemm_bias(
    const unsigned short* __restrict__ A, const unsigned short* __restrict__ Bw,
    const float* __restrict__ bias, void* __restrict__ Out) {
  __shared__ unsigned short As[2][128 * 40];   // 32 K cols + 8 pad (bank-safe)
  __shared__ unsigned short Bs[2][128 * 40];

  const int tid    = threadIdx.x;
  const int lane   = tid & 31;
  const int w      = tid >> 5;
  const int wm     = w >> 2;                // 0..1 -> 64 rows
  const int wn     = w & 3;                 // 0..3 -> 32 cols
  const int lane15 = lane & 15;
  const int kkA    = (lane & 16) ? 8 : 0;   // A: 8-wide K sub-blocks per half
  const int koB    = (lane & 16) ? 8 : 0;   // B: 8-wide K sub-blocks per half
  const int rhalf  = (lane >> 4) & 1;
  const int m0     = blockIdx.x * 128;
  const int n0     = blockIdx.y * 128;

  auto stage = [&](int k0, int buf) {
    for (int i = tid; i < 128 * 4; i += 256) {
      const int r = i >> 2, c = i & 3;
      async_cp16(A + (size_t)(m0 + r) * DD + k0 + c * 8, &As[buf][r * 40 + c * 8]);
    }
    for (int i = tid; i < 128 * 4; i += 256) {
      const int r = i >> 2, c = i & 3;
      async_cp16(Bw + (size_t)(n0 + r) * DD + k0 + c * 8, &Bs[buf][r * 40 + c * 8]);
    }
  };

  const v8f vz = {0.f, 0.f, 0.f, 0.f, 0.f, 0.f, 0.f, 0.f};
  v8f acc[4][2];
#pragma unroll
  for (int t = 0; t < 4; ++t)
#pragma unroll
    for (int n = 0; n < 2; ++n) acc[t][n] = vz;

  stage(0, 0);
  async_wait_all();
  __syncthreads();

  for (int k0 = 0; k0 < DD; k0 += 32) {
    const int buf = (k0 >> 5) & 1;
    if (k0 + 32 < DD) stage(k0 + 32, buf ^ 1);   // prefetch next tile

    FragAB a[4], b[2];
#pragma unroll
    for (int t = 0; t < 4; ++t) {
      const unsigned short* p = &As[buf][(wm * 64 + t * 16 + lane15) * 40];
      a[t].u[0] = *(const usv8*)(p + kkA);
      a[t].u[1] = *(const usv8*)(p + 16 + kkA);
    }
#pragma unroll
    for (int n = 0; n < 2; ++n) {
      const unsigned short* p = &Bs[buf][(wn * 32 + n * 16 + lane15) * 40];
      b[n].u[0] = *(const usv8*)(p + koB);
      b[n].u[1] = *(const usv8*)(p + koB + 16);
    }
#pragma unroll
    for (int t = 0; t < 4; ++t)
#pragma unroll
      for (int n = 0; n < 2; ++n) acc[t][n] = wmma_bf16(a[t], b[n], acc[t][n]);

    async_wait_all();
    __syncthreads();
  }

#pragma unroll
  for (int n = 0; n < 2; ++n) {
    const int ncol = n0 + wn * 32 + n * 16 + lane15;
    const float bv = bias[ncol];
#pragma unroll
    for (int t = 0; t < 4; ++t) {
      const int mbase = m0 + wm * 64 + t * 16 + rhalf * 8;
#pragma unroll
      for (int v = 0; v < 8; ++v) {
        const float val = acc[t][n][v] + bv;
        if constexpr (OUT_F32)
          ((float*)Out)[(size_t)(mbase + v) * DD + ncol] = val;
        else
          ((unsigned short*)Out)[(size_t)(mbase + v) * DD + ncol] = f2bf(val);
      }
    }
  }
}

// ---------------------------------------------------------------------------
// 3) V[B,N,D] -> Vt[B,D,N]  (LDS tiled transpose, bf16)
// ---------------------------------------------------------------------------
__global__ void transpose_bf(const unsigned short* __restrict__ V,
                             unsigned short* __restrict__ Vt) {
  __shared__ unsigned short tile[32][33];
  const int b  = blockIdx.z;
  const int n0 = blockIdx.y * 32;
  const int d0 = blockIdx.x * 32;
  const int tx = threadIdx.x, ty = threadIdx.y;
#pragma unroll
  for (int i = 0; i < 4; ++i) {
    const int r = ty + i * 8;
    tile[r][tx] = V[((size_t)(b * BN + n0 + r)) * DD + d0 + tx];
  }
  __syncthreads();
#pragma unroll
  for (int i = 0; i < 4; ++i) {
    const int r = ty + i * 8;
    Vt[((size_t)(b * DD + d0 + r)) * BN + n0 + tx] = tile[tx][r];
  }
}

// ---------------------------------------------------------------------------
// 4) Attention pass 1: row max m and denom l of masked softmax(Q K^T)
//    block = 64 queries, 8 waves = 4(q-tile) x 2(key-half),
//    double-buffered 64-key LDS tiles, async staging, dual WMMA accumulators
// ---------------------------------------------------------------------------
__global__ __launch_bounds__(256) void attn_pass1(
    const unsigned short* __restrict__ Q, const unsigned short* __restrict__ K,
    const int* __restrict__ mask, float* __restrict__ mOut,
    float* __restrict__ lOut) {
  extern __shared__ char smem_raw[];
  unsigned short* Qs  = (unsigned short*)smem_raw;  // 64 x 776
  unsigned short* Ks0 = Qs + 64 * 776;              // 2 x (64 x 776)
  float* comb = (float*)(Ks0 + 2 * 64 * 776);       // [2][64][2]

  const int tid    = threadIdx.x;
  const int b      = blockIdx.y;
  const int q0     = blockIdx.x * 64;
  const int lane   = tid & 31;
  const int w      = tid >> 5;
  const int wq     = w >> 1;   // 0..3
  const int wk     = w & 1;    // 0..1
  const int lane15 = lane & 15;
  const int kkA    = (lane & 16) ? 8 : 0;
  const int koB    = (lane & 16) ? 8 : 0;
  const int rhalf  = (lane >> 4) & 1;

  auto stageK = [&](int kb, int buf) {
    unsigned short* dst = Ks0 + buf * (64 * 776);
    for (int i = tid; i < 64 * 96; i += 256) {
      const int r = i / 96, c = i % 96;
      async_cp16(K + ((size_t)(b * BN + kb + r)) * DD + c * 8, &dst[r * 776 + c * 8]);
    }
  };

  for (int i = tid; i < 64 * 96; i += 256) {
    const int r = i / 96, c = i % 96;
    async_cp16(Q + ((size_t)(b * BN + q0 + r)) * DD + c * 8, &Qs[r * 776 + c * 8]);
  }
  stageK(0, 0);
  async_wait_all();
  __syncthreads();

  float mrun[8], lrun[8];
#pragma unroll
  for (int v = 0; v < 8; ++v) { mrun[v] = -3.0e38f; lrun[v] = 0.f; }

  const v8f vz = {0.f, 0.f, 0.f, 0.f, 0.f, 0.f, 0.f, 0.f};

  for (int kb = 0; kb < BN; kb += 64) {
    const int buf = (kb >> 6) & 1;
    if (kb + 64 < BN) stageK(kb + 64, buf ^ 1);    // prefetch next key tile
    const unsigned short* Ksb = Ks0 + buf * (64 * 776);

#pragma unroll
    for (int kt = 0; kt < 2; ++kt) {
      const int ktile = wk * 2 + kt;
      v8f acc0 = vz, acc1 = vz;                    // two independent chains
      const unsigned short* arow = &Qs[(wq * 16 + lane15) * 776];
      const unsigned short* brow = &Ksb[(ktile * 16 + lane15) * 776];
#pragma unroll 3
      for (int d0 = 0; d0 < DD; d0 += 64) {
        FragAB a0, b0, a1, b1;
        a0.u[0] = *(const usv8*)(arow + d0 + kkA);
        a0.u[1] = *(const usv8*)(arow + d0 + 16 + kkA);
        b0.u[0] = *(const usv8*)(brow + d0 + koB);
        b0.u[1] = *(const usv8*)(brow + d0 + koB + 16);
        a1.u[0] = *(const usv8*)(arow + d0 + 32 + kkA);
        a1.u[1] = *(const usv8*)(arow + d0 + 48 + kkA);
        b1.u[0] = *(const usv8*)(brow + d0 + 32 + koB);
        b1.u[1] = *(const usv8*)(brow + d0 + 48 + koB);
        acc0 = wmma_bf16(a0, b0, acc0);
        acc1 = wmma_bf16(a1, b1, acc1);
      }
      const v8f acc = acc0 + acc1;

      const int key = kb + ktile * 16 + lane15;    // this lane's key column
      const int mk  = mask[b * BN + key];
#pragma unroll
      for (int v = 0; v < 8; ++v) {
        const float s = (mk == 0) ? -1.0e30f : acc[v];
        float x = s;                               // 16-lane row-group max
        x = fmaxf(x, __shfl_xor(x, 1, 32));
        x = fmaxf(x, __shfl_xor(x, 2, 32));
        x = fmaxf(x, __shfl_xor(x, 4, 32));
        x = fmaxf(x, __shfl_xor(x, 8, 32));
        const float mnew = fmaxf(mrun[v], x);
        float e = (mk == 0) ? 0.f : __expf(s - mnew);
        e += __shfl_xor(e, 1, 32);
        e += __shfl_xor(e, 2, 32);
        e += __shfl_xor(e, 4, 32);
        e += __shfl_xor(e, 8, 32);
        lrun[v] = lrun[v] * __expf(mrun[v] - mnew) + e;
        mrun[v] = mnew;
      }
    }
    async_wait_all();     // next key tile landed
    __syncthreads();
  }

  if (lane15 == 0) {   // lanes 0 and 16 each own 8 rows
#pragma unroll
    for (int v = 0; v < 8; ++v) {
      const int row = wq * 16 + rhalf * 8 + v;
      comb[(wk * 64 + row) * 2 + 0] = mrun[v];
      comb[(wk * 64 + row) * 2 + 1] = lrun[v];
    }
  }
  __syncthreads();
  for (int r = tid; r < 64; r += 256) {
    const float m0v = comb[r * 2 + 0],        l0v = comb[r * 2 + 1];
    const float m1v = comb[(64 + r) * 2 + 0], l1v = comb[(64 + r) * 2 + 1];
    const float mm = fmaxf(m0v, m1v);
    const float ll = l0v * __expf(m0v - mm) + l1v * __expf(m1v - mm);
    mOut[b * BN + q0 + r] = mm;
    lOut[b * BN + q0 + r] = ll;
  }
}

// ---------------------------------------------------------------------------
// 5) Attention pass 2: merged = (softmax(QK^T) @ V) * qmask  (bf16 out)
//    block = 64 queries, 8 waves = 4(q-tile) x 2(d-half of 384),
//    32-key tiles, double-buffered K, async staging
// ---------------------------------------------------------------------------
__global__ __launch_bounds__(256) void attn_pass2(
    const unsigned short* __restrict__ Q, const unsigned short* __restrict__ K,
    const unsigned short* __restrict__ Vt, const int* __restrict__ mask,
    const float* __restrict__ mIn, const float* __restrict__ lIn,
    unsigned short* __restrict__ Merged) {
  extern __shared__ char smem_raw[];
  unsigned short* Qs  = (unsigned short*)smem_raw;  // 64 x 776
  unsigned short* Ks0 = Qs + 64 * 776;              // 2 x (32 x 776)
  unsigned short* Vts = Ks0 + 2 * 32 * 776;         // 768 x 40 (d-major, 32 keys)
  unsigned short* Ps  = Vts + 768 * 40;             // 64 x 40  (q-major, 32 keys)
  float* mls = (float*)(Ps + 64 * 40);              // 64
  float* lls = mls + 64;                            // 64

  const int tid    = threadIdx.x;
  const int b      = blockIdx.y;
  const int q0     = blockIdx.x * 64;
  const int lane   = tid & 31;
  const int w      = tid >> 5;
  const int wq     = w >> 1;   // 0..3 : q tile
  const int wd     = w & 1;    // 0..1 : 384-wide d half, also its S k-tile
  const int lane15 = lane & 15;
  const int kkA    = (lane & 16) ? 8 : 0;
  const int koB    = (lane & 16) ? 8 : 0;
  const int rhalf  = (lane >> 4) & 1;

  auto stageK = [&](int kb, int buf) {
    unsigned short* dst = Ks0 + buf * (32 * 776);
    for (int i = tid; i < 32 * 96; i += 256) {
      const int r = i / 96, c = i % 96;
      async_cp16(K + ((size_t)(b * BN + kb + r)) * DD + c * 8, &dst[r * 776 + c * 8]);
    }
  };
  auto stageV = [&](int kb) {
    for (int i = tid; i < 768 * 4; i += 256) {
      const int r = i >> 2, c = i & 3;
      async_cp16(Vt + ((size_t)(b * DD + r)) * BN + kb + c * 8, &Vts[r * 40 + c * 8]);
    }
  };

  for (int i = tid; i < 64 * 96; i += 256) {
    const int r = i / 96, c = i % 96;
    async_cp16(Q + ((size_t)(b * BN + q0 + r)) * DD + c * 8, &Qs[r * 776 + c * 8]);
  }
  for (int r = tid; r < 64; r += 256) {
    mls[r] = mIn[b * BN + q0 + r];
    lls[r] = lIn[b * BN + q0 + r];
  }
  stageK(0, 0);
  async_wait_all();
  __syncthreads();

  const v8f vz = {0.f, 0.f, 0.f, 0.f, 0.f, 0.f, 0.f, 0.f};
  v8f o[24];
#pragma unroll
  for (int t = 0; t < 24; ++t) o[t] = vz;

  float qmf[8];
#pragma unroll
  for (int v = 0; v < 8; ++v)
    qmf[v] = (float)mask[b * BN + q0 + wq * 16 + rhalf * 8 + v];

  for (int kb = 0; kb < BN; kb += 32) {
    const int buf = (kb >> 5) & 1;
    if (kb + 32 < BN) stageK(kb + 32, buf ^ 1);   // prefetch next K tile
    stageV(kb);                                   // this iteration's V tile
    const unsigned short* Ksb = Ks0 + buf * (32 * 776);

    // S tile [16q x 16k], this wave's k-tile = wd  (two independent chains)
    v8f s0 = vz, s1 = vz;
    {
      const unsigned short* arow = &Qs[(wq * 16 + lane15) * 776];
      const unsigned short* brow = &Ksb[(wd * 16 + lane15) * 776];
#pragma unroll 3
      for (int d0 = 0; d0 < DD; d0 += 64) {
        FragAB a0, b0, a1, b1;
        a0.u[0] = *(const usv8*)(arow + d0 + kkA);
        a0.u[1] = *(const usv8*)(arow + d0 + 16 + kkA);
        b0.u[0] = *(const usv8*)(brow + d0 + koB);
        b0.u[1] = *(const usv8*)(brow + d0 + koB + 16);
        a1.u[0] = *(const usv8*)(arow + d0 + 32 + kkA);
        a1.u[1] = *(const usv8*)(arow + d0 + 48 + kkA);
        b1.u[0] = *(const usv8*)(brow + d0 + 32 + koB);
        b1.u[1] = *(const usv8*)(brow + d0 + 48 + koB);
        s0 = wmma_bf16(a0, b0, s0);
        s1 = wmma_bf16(a1, b1, s1);
      }
    }
    const v8f s = s0 + s1;

    const int key = kb + wd * 16 + lane15;
    const int mk  = mask[b * BN + key];
#pragma unroll
    for (int v = 0; v < 8; ++v) {
      const int row = wq * 16 + rhalf * 8 + v;
      const float p = (mk == 0) ? 0.f : __expf(s[v] - mls[row]) / lls[row];
      Ps[row * 40 + wd * 16 + lane15] = f2bf(p);
    }
    async_wait_all();   // V tile (and next K tile) landed
    __syncthreads();    // sibling wave's P half + V tile visible

    FragAB pa;
    const unsigned short* prow = &Ps[(wq * 16 + lane15) * 40];
    pa.u[0] = *(const usv8*)(prow + kkA);
    pa.u[1] = *(const usv8*)(prow + 16 + kkA);
#pragma unroll
    for (int t = 0; t < 24; ++t) {
      const unsigned short* vrow = &Vts[(wd * 384 + t * 16 + lane15) * 40];
      FragAB vb;
      vb.u[0] = *(const usv8*)(vrow + koB);
      vb.u[1] = *(const usv8*)(vrow + koB + 16);
      o[t] = wmma_bf16(pa, vb, o[t]);
    }
    __syncthreads();    // PV reads done before next iteration restages Vts/Ps
  }

#pragma unroll
  for (int t = 0; t < 24; ++t) {
    const int dcol = wd * 384 + t * 16 + lane15;
#pragma unroll
    for (int v = 0; v < 8; ++v) {
      const int qrow = q0 + wq * 16 + rhalf * 8 + v;
      Merged[((size_t)(b * BN + qrow)) * DD + dcol] = f2bf(o[t][v] * qmf[v]);
    }
  }
}

// ---------------------------------------------------------------------------
// Host-side orchestration
// ---------------------------------------------------------------------------
extern "C" void kernel_launch(void* const* d_in, const int* in_sizes, int n_in,
                              void* d_out, int out_size, void* d_ws,
                              size_t ws_size, hipStream_t stream) {
  (void)in_sizes; (void)n_in; (void)out_size; (void)ws_size;

  const float* query = (const float*)d_in[0];
  const float* value = (const float*)d_in[1];
  const int*   mask  = (const int*)d_in[2];
  const float* q_w  = (const float*)d_in[3];
  const float* q_b  = (const float*)d_in[4];
  const float* k_w  = (const float*)d_in[5];
  const float* k_b  = (const float*)d_in[6];
  const float* v_w  = (const float*)d_in[7];
  const float* v_b  = (const float*)d_in[8];
  const float* fc_w = (const float*)d_in[9];
  const float* fc_b = (const float*)d_in[10];
  float* out = (float*)d_out;

  const size_t WSZ = (size_t)DD * DD;        // 589,824
  const size_t TSZ = (size_t)NB * BN * DD;   // 12,582,912

  unsigned short* ws     = (unsigned short*)d_ws;
  unsigned short* wq_bf  = ws;
  unsigned short* wk_bf  = wq_bf + WSZ;
  unsigned short* wv_bf  = wk_bf + WSZ;
  unsigned short* wfc_bf = wv_bf + WSZ;
  unsigned short* Xq     = wfc_bf + WSZ;   // query bf16
  unsigned short* Xv     = Xq + TSZ;       // value bf16
  unsigned short* Qbf    = Xv + TSZ;
  unsigned short* Kbf    = Qbf + TSZ;
  unsigned short* Vbf    = Kbf + TSZ;
  unsigned short* Vtb    = Vbf + TSZ;
  unsigned short* Mrg    = Vtb + TSZ;
  float* mbuf = (float*)(Mrg + TSZ);
  float* lbuf = mbuf + (size_t)NB * BN;

  // 1) weights + activations -> bf16
  {
    dim3 blk(256);
    dim3 gw((unsigned)((WSZ / 4 + 255) / 256));
    cvt_f32_bf16<<<gw, blk, 0, stream>>>(q_w,  wq_bf,  (int)WSZ);
    cvt_f32_bf16<<<gw, blk, 0, stream>>>(k_w,  wk_bf,  (int)WSZ);
    cvt_f32_bf16<<<gw, blk, 0, stream>>>(v_w,  wv_bf,  (int)WSZ);
    cvt_f32_bf16<<<gw, blk, 0, stream>>>(fc_w, wfc_bf, (int)WSZ);
    dim3 gx((unsigned)((TSZ / 4 + 255) / 256));
    cvt_f32_bf16<<<gx, blk, 0, stream>>>(query, Xq, (int)TSZ);
    cvt_f32_bf16<<<gx, blk, 0, stream>>>(value, Xv, (int)TSZ);
  }

  const int M = NB * BN;  // 16384
  // 2) Q/K/V projections (bf16 A, bf16 out)
  {
    dim3 g(M / 128, DD / 128), blk(256);
    gemm_bias<false><<<g, blk, 0, stream>>>(Xq, wq_bf, q_b, Qbf);
    gemm_bias<false><<<g, blk, 0, stream>>>(Xv, wk_bf, k_b, Kbf);
    gemm_bias<false><<<g, blk, 0, stream>>>(Xv, wv_bf, v_b, Vbf);
  }
  // 3) V -> V^T
  {
    dim3 g(DD / 32, BN / 32, NB), blk(32, 8);
    transpose_bf<<<g, blk, 0, stream>>>(Vbf, Vtb);
  }
  // 4) softmax statistics
  {
    const int shmem = (64 * 776 * 3) * 2 + 2 * 64 * 2 * 4;  // 299,008 B
    (void)hipFuncSetAttribute((const void*)attn_pass1,
                              hipFuncAttributeMaxDynamicSharedMemorySize, shmem);
    dim3 g(BN / 64, NB), blk(256);
    attn_pass1<<<g, blk, shmem, stream>>>(Qbf, Kbf, mask, mbuf, lbuf);
  }
  // 5) attention output (merged, bf16)
  {
    const int shmem =
        (64 * 776 + 2 * 32 * 776 + 768 * 40 + 64 * 40) * 2 + 128 * 4;  // 265,728 B
    (void)hipFuncSetAttribute((const void*)attn_pass2,
                              hipFuncAttributeMaxDynamicSharedMemorySize, shmem);
    dim3 g(BN / 64, NB), blk(256);
    attn_pass2<<<g, blk, shmem, stream>>>(Qbf, Kbf, Vtb, mask, mbuf, lbuf, Mrg);
  }
  // 6) final FC (bf16 A, fp32 out)
  {
    dim3 g(M / 128, DD / 128), blk(256);
    gemm_bias<true><<<g, blk, 0, stream>>>(Mrg, wfc_bf, fc_b, out);
  }
}
